// Stage2BeamModel_57655640982186
// MI455X (gfx1250) — compile-verified, matrix operand order checked by reading, not run on hardware
//
#include <hip/hip_runtime.h>
#include <hip/hip_bf16.h>
#include <cstdint>
#include <cstddef>

// ---------------------------------------------------------------------------
// Types for CDNA5 WMMA (wave32): 16x16x32 bf16 -> f32 accumulate
// ---------------------------------------------------------------------------
typedef __attribute__((ext_vector_type(16))) __bf16 bf16x16;
typedef __attribute__((ext_vector_type(8)))  float  f32x8;

union FragAB {
  bf16x16 v;
  uint4   q[2];
};

#define HID 128   // fan_out of both SAGE layers

// ---------------------------------------------------------------------------
// Utility: zero an f32 buffer (workspace is poisoned 0xAA by the harness)
// ---------------------------------------------------------------------------
__global__ void zero_f32(float* __restrict__ p, long long n) {
  long long i = (long long)blockIdx.x * blockDim.x + threadIdx.x;
  long long stride = (long long)gridDim.x * blockDim.x;
  for (; i < n; i += stride) p[i] = 0.0f;
}

// ---------------------------------------------------------------------------
// Degree count: atomic +1.0 per edge at dst (resolves in L2)
// ---------------------------------------------------------------------------
__global__ void deg_count(const int* __restrict__ dst, float* __restrict__ deg,
                          int nE) {
  int e = blockIdx.x * blockDim.x + threadIdx.x;
  if (e < nE) atomicAdd(&deg[dst[e]], 1.0f);
}

// ---------------------------------------------------------------------------
// Edge scatter-add: one thread per (edge, 4 features). float4 gather from src
// row, 4 global_atomic_add_f32 into dst row. Feature tables are L2-resident.
// ---------------------------------------------------------------------------
__global__ void scatter_add(const float* __restrict__ feat,
                            const int* __restrict__ src,
                            const int* __restrict__ dst,
                            float* __restrict__ agg,
                            int nE, int F) {
  int tid = blockIdx.x * blockDim.x + threadIdx.x;
  int fq  = F >> 2;
  int e   = tid / fq;
  if (e >= nE) return;
  int j = (tid - e * fq) << 2;
  int s = src[e], d = dst[e];
  float4 v = *(const float4*)(feat + (size_t)s * F + j);
  float* dp = agg + (size_t)d * F + j;
  atomicAdd(dp + 0, v.x);
  atomicAdd(dp + 1, v.y);
  atomicAdd(dp + 2, v.z);
  atomicAdd(dp + 3, v.w);
}

// ---------------------------------------------------------------------------
// Build bf16 A matrix  [N x 2F] = [ agg/max(deg,1) | root ]
// ---------------------------------------------------------------------------
__global__ void build_acat(const float* __restrict__ agg,
                           const float* __restrict__ deg,
                           const float* __restrict__ root,
                           __bf16* __restrict__ Acat,
                           int nrows, int F) {
  const int W = 2 * F;
  long long tid = (long long)blockIdx.x * blockDim.x + threadIdx.x;
  long long total = (long long)nrows * W;
  if (tid >= total) return;
  int i = (int)(tid / W);
  int j = (int)(tid - (long long)i * W);
  float v;
  if (j < F) v = agg[(size_t)i * F + j] / fmaxf(deg[i], 1.0f);
  else       v = root[(size_t)i * F + (j - F)];
  Acat[tid] = (__bf16)v;
}

// ---------------------------------------------------------------------------
// Swizzle weights into WMMA B-fragment order.
// Wcat[k][n] : k<F -> Wl[k][n] (agg path), k>=F -> Wr[k-F][n] (root path).
// B-fragment (16-bit, 32x16): lane = {half=lane>>4, n=lane&15},
//   element e: k_local = (e>>3)*16 + half*8 + (e&7).
// Buffer order: ((kt*8 + nt)*32 + lane)*16 + e  -> fragment load is 2x b128.
// ---------------------------------------------------------------------------
__global__ void swizzle_w(const float* __restrict__ Wl,
                          const float* __restrict__ Wr,
                          __bf16* __restrict__ Wsw, int F) {
  const int K = 2 * F;
  int tid = blockIdx.x * blockDim.x + threadIdx.x;
  if (tid >= K * HID) return;
  int e    = tid & 15;
  int lane = (tid >> 4) & 31;
  int frag = tid >> 9;            // kt*8 + nt
  int kt = frag >> 3, nt = frag & 7;
  int k = kt * 32 + ((e >> 3) << 4) + ((lane >> 4) << 3) + (e & 7);
  int n = nt * 16 + (lane & 15);
  float v = (k < F) ? Wl[k * HID + n] : Wr[(k - F) * HID + n];
  Wsw[tid] = (__bf16)v;
}

// ---------------------------------------------------------------------------
// WMMA GEMM: out[r][c] = relu(sum_k A[r][k]*Wcat[k][c] + bias[c])
// 256 threads = 8 waves; block owns 128 rows; wave owns 16 rows x 128 cols
// = 8 C tiles. Weights staged in dynamic LDS in fragment order.
// ---------------------------------------------------------------------------
__global__ void gemm_sage_wmma(const __bf16* __restrict__ A,
                               const __bf16* __restrict__ Wsw,
                               const float* __restrict__ bias,
                               float* __restrict__ out,
                               int nrows, int K) {
  extern __shared__ char smem[];
  __bf16* sW = (__bf16*)smem;
  {
    const uint4* s = (const uint4*)Wsw;
    uint4* d = (uint4*)smem;
    const int nChunks = (K * HID) >> 3;          // 16-byte chunks
    for (int i = threadIdx.x; i < nChunks; i += blockDim.x) d[i] = s[i];
  }
  __syncthreads();

  const int wave = threadIdx.x >> 5;
  const int lane = threadIdx.x & 31;
  const int half = lane >> 4;
  const int l15  = lane & 15;
  const int m0   = blockIdx.x * 128 + wave * 16;

  int r = m0 + l15;
  if (r >= nrows) r = nrows - 1;                 // clamp: keep EXEC all-ones
  const __bf16* arow = A + (size_t)r * K;

  f32x8 c[8] = {};

  const int kTiles = K >> 5;
  for (int kt = 0; kt < kTiles; ++kt) {
    const int kk = kt << 5;
    FragAB a;
    a.q[0] = *(const uint4*)(arow + kk + half * 8);
    a.q[1] = *(const uint4*)(arow + kk + 16 + half * 8);
#pragma unroll
    for (int nt = 0; nt < 8; ++nt) {
      FragAB b;
      const __bf16* bp = sW + ((((kt << 3) + nt) << 5) + lane) * 16;
      b.q[0] = *(const uint4*)(bp);
      b.q[1] = *(const uint4*)(bp + 8);
      c[nt] = __builtin_amdgcn_wmma_f32_16x16x32_bf16(
          false, a.v, false, b.v, (short)0, c[nt], false, false);
    }
  }

#pragma unroll
  for (int nt = 0; nt < 8; ++nt) {
    const int col = (nt << 4) + l15;
    const float bv = bias[col];
#pragma unroll
    for (int v = 0; v < 8; ++v) {
      const int row = m0 + v + (half << 3);
      if (row < nrows) {
        float val = c[nt][v] + bv;
        out[(size_t)row * HID + col] = val > 0.0f ? val : 0.0f;
      }
    }
  }
}

// ---------------------------------------------------------------------------
// Output heads: [N x 128] @ ([128x3] | [128x8]) + biases, weights in LDS.
// ---------------------------------------------------------------------------
__global__ void heads_kernel(const float* __restrict__ h,
                             const float* __restrict__ Wh,
                             const float* __restrict__ bh,
                             const float* __restrict__ Wm,
                             const float* __restrict__ bm,
                             float* __restrict__ out, int nrows) {
  __shared__ float sW[HID * 11 + 16];
  for (int i = threadIdx.x; i < HID * 11; i += blockDim.x) {
    int k = i / 11, c = i - k * 11;
    sW[i] = (c < 3) ? Wh[k * 3 + c] : Wm[k * 8 + (c - 3)];
  }
  for (int i = threadIdx.x; i < 11; i += blockDim.x)
    sW[HID * 11 + i] = (i < 3) ? bh[i] : bm[i - 3];
  __syncthreads();

  int node = blockIdx.x * blockDim.x + threadIdx.x;
  if (node >= nrows) return;
  const float* hr = h + (size_t)node * HID;
  float acc[11];
#pragma unroll
  for (int c = 0; c < 11; ++c) acc[c] = sW[HID * 11 + c];
  for (int k = 0; k < HID; ++k) {
    float hv = hr[k];
#pragma unroll
    for (int c = 0; c < 11; ++c) acc[c] += hv * sW[k * 11 + c];
  }
#pragma unroll
  for (int c = 0; c < 3; ++c) out[(size_t)node * 3 + c] = acc[c];
#pragma unroll
  for (int c = 0; c < 8; ++c)
    out[(size_t)nrows * 3 + (size_t)node * 8 + c] = acc[3 + c];
}

// ---------------------------------------------------------------------------
// Host-side launch
// ---------------------------------------------------------------------------
extern "C" void kernel_launch(void* const* d_in, const int* in_sizes, int n_in,
                              void* d_out, int out_size, void* d_ws, size_t ws_size,
                              hipStream_t stream) {
  const float* x   = (const float*)d_in[0];
  const int*   ei  = (const int*)d_in[1];
  const float* Wl1 = (const float*)d_in[2];
  const float* Wr1 = (const float*)d_in[3];
  const float* b1  = (const float*)d_in[4];
  const float* Wl2 = (const float*)d_in[5];
  const float* Wr2 = (const float*)d_in[6];
  const float* b2  = (const float*)d_in[7];
  const float* Wh  = (const float*)d_in[8];
  const float* bh  = (const float*)d_in[9];
  const float* Wm  = (const float*)d_in[10];
  const float* bm  = (const float*)d_in[11];
  float* out = (float*)d_out;

  const int N = in_sizes[0] / 64;     // 100000 nodes, IN_DIM = 64
  const int E = in_sizes[1] / 2;      // 1.6M edges
  const int* src = ei;
  const int* dst = ei + E;

  // Workspace carve-out (256B aligned regions)
  size_t off = 0;
  auto carve = [&](size_t bytes) -> void* {
    void* p = (char*)d_ws + off;
    off += (bytes + 255) & ~(size_t)255;
    return p;
  };
  float*  deg  = (float*)carve((size_t)N * 4);
  float*  agg  = (float*)carve((size_t)N * HID * 4);
  float*  h    = (float*)carve((size_t)N * HID * 4);
  __bf16* Acat = (__bf16*)carve((size_t)N * 2 * HID * 2);
  __bf16* Wsw1 = (__bf16*)carve((size_t)128 * HID * 2);
  __bf16* Wsw2 = (__bf16*)carve((size_t)256 * HID * 2);

  const int T = 256;
  const int gemmBlocks = (N + 127) / 128;

  // ---- Layer 1 (F = 64) ----
  zero_f32<<<2048, T, 0, stream>>>(deg, N);
  zero_f32<<<4096, T, 0, stream>>>(agg, (long long)N * 64);
  deg_count<<<(E + T - 1) / T, T, 0, stream>>>(dst, deg, E);
  swizzle_w<<<(128 * HID + T - 1) / T, T, 0, stream>>>(Wl1, Wr1, Wsw1, 64);
  {
    long long work = (long long)E * 16;   // edge x (64/4)
    scatter_add<<<(unsigned)((work + T - 1) / T), T, 0, stream>>>(x, src, dst, agg, E, 64);
  }
  {
    long long work = (long long)N * 128;
    build_acat<<<(unsigned)((work + T - 1) / T), T, 0, stream>>>(agg, deg, x, Acat, N, 64);
  }
  gemm_sage_wmma<<<gemmBlocks, T, 128 * HID * 2, stream>>>(Acat, Wsw1, b1, h, N, 128);

  // ---- Layer 2 (F = 128) ----
  zero_f32<<<4096, T, 0, stream>>>(agg, (long long)N * HID);
  swizzle_w<<<(256 * HID + T - 1) / T, T, 0, stream>>>(Wl2, Wr2, Wsw2, 128);
  {
    long long work = (long long)E * 32;   // edge x (128/4)
    scatter_add<<<(unsigned)((work + T - 1) / T), T, 0, stream>>>(h, src, dst, agg, E, 128);
  }
  {
    long long work = (long long)N * 256;
    build_acat<<<(unsigned)((work + T - 1) / T), T, 0, stream>>>(agg, deg, h, Acat, N, 128);
  }
  gemm_sage_wmma<<<gemmBlocks, T, 256 * HID * 2, stream>>>(Acat, Wsw2, b2, h, N, 256);

  // ---- Heads ----
  heads_kernel<<<(N + T - 1) / T, T, 0, stream>>>(h, Wh, bh, Wm, bm, out, N);
}
